// CRF_85538568667675
// MI455X (gfx1250) — matrix operand order repaired and checked by reading, not compiled, not run on hardware
//
#include <hip/hip_runtime.h>

typedef __attribute__((ext_vector_type(16))) __bf16 v16bf;
typedef __attribute__((ext_vector_type(8)))  float  v8f;

#define Bn 256
#define Tn 1024
#define Ln 64

// ---------------------------------------------------------------------------
// Kernel 1: P[b,t,:] = exp(logits[b,t,:] - rowmax) as bf16, c[b,t] = rowmax.
// One wave (32 lanes) per (b,t) row; lane holds 2 consecutive elements.
// ---------------------------------------------------------------------------
__global__ void k_exp_rows(const float* __restrict__ logits,
                           __bf16* __restrict__ P, float* __restrict__ c) {
    const int row = blockIdx.x;          // b*T + t
    const int l   = threadIdx.x;         // 0..31
    const float2 v = reinterpret_cast<const float2*>(logits + (long)row * Ln)[l];
    float m = fmaxf(v.x, v.y);
    for (int off = 16; off > 0; off >>= 1) m = fmaxf(m, __shfl_xor(m, off));
    union { unsigned int u; __bf16 b[2]; } pk;
    pk.b[0] = (__bf16)__expf(v.x - m);
    pk.b[1] = (__bf16)__expf(v.y - m);
    reinterpret_cast<unsigned int*>(P + (long)row * Ln)[l] = pk.u;
    if (l == 0) c[row] = m;
}

// ---------------------------------------------------------------------------
// Kernel 2: E = exp(trans) as bf16 (64x64, row-major; K=i rows, N=j cols)
// ---------------------------------------------------------------------------
__global__ void k_exp_trans(const float* __restrict__ trans, __bf16* __restrict__ E) {
    for (int i = threadIdx.x; i < Ln * Ln; i += blockDim.x)
        E[i] = (__bf16)__expf(trans[i]);
}

// ---------------------------------------------------------------------------
// Kernel 3: masked unary + pairwise sequence score per batch
// ---------------------------------------------------------------------------
__global__ void k_seq(const float* __restrict__ logits, const int* __restrict__ labels,
                      const int* __restrict__ lens, const float* __restrict__ trans,
                      float* __restrict__ seq) {
    const int b = blockIdx.x;
    const int len = lens[b];
    float acc = 0.f;
    for (int t = threadIdx.x; t < Tn; t += blockDim.x) {
        if (t < len) {
            const int lab = labels[b * Tn + t];
            acc += logits[((long)(b * Tn + t)) * Ln + lab];
            if (t >= 1) acc += trans[labels[b * Tn + t - 1] * Ln + lab];
        }
    }
    __shared__ float red[256];
    red[threadIdx.x] = acc; __syncthreads();
    for (int s = 128; s > 0; s >>= 1) {
        if ((int)threadIdx.x < s) red[threadIdx.x] += red[threadIdx.x + s];
        __syncthreads();
    }
    if (threadIdx.x == 0) seq[b] = red[0];
}

// ---------------------------------------------------------------------------
// Kernel 4: forward scan. One wave per 16-batch tile.
//   state a[M][j] (scaled exp domain), E resident in VGPRs as 8 WMMA-B frags.
//   per step: a(bf16)->LDS -> A frags -> 8x v_wmma_f32_16x16x32_bf16 ->
//   D scatter to LDS -> row gather -> *P, renorm, mask, log-accumulate.
// ---------------------------------------------------------------------------
#define RSA 36   // lA row stride in dwords (16B-aligned rows)
#define RSS 68   // lS row stride in floats (16B-aligned rows)

__global__ void __launch_bounds__(32, 1)
k_scan(const __bf16* __restrict__ P, const float* __restrict__ c,
       const __bf16* __restrict__ E, const int* __restrict__ lens,
       float* __restrict__ lognorm) {
    const int l    = threadIdx.x;
    const int M    = l & 15;        // batch row within tile (also WMMA A/M role)
    const int half = l >> 4;        // 0/1
    const int b    = blockIdx.x * 16 + M;

    // --- persistent WMMA-B fragments of E: lane holds N = nt*16 + l%16, K = kc*32 + table order (8h+s | 16+8h+s-8)
    v16bf Bf[2][4];
    for (int kc = 0; kc < 2; ++kc)
        for (int nt = 0; nt < 4; ++nt) {
            v16bf tmp;
            for (int s = 0; s < 16; ++s)
                tmp[s] = E[(kc * 32 + half * 8 + (s < 8 ? s : 8 + s)) * Ln + nt * 16 + M];
            Bf[kc][nt] = tmp;
        }

    const int len = lens[b];

    // --- init: a = P[b,0,:]  (lane holds j = half*32 + q), logAcc = c[b,0]
    float a[32];
    {
        const uint4* src = reinterpret_cast<const uint4*>(P + ((long)b * Tn) * Ln + half * 32);
        for (int cc = 0; cc < 4; ++cc) {
            union { uint4 u; __bf16 bv[8]; } u8; u8.u = src[cc];
            for (int e = 0; e < 8; ++e) a[cc * 8 + e] = (float)u8.bv[e];
        }
    }
    float logAcc = c[(long)b * Tn];

    __shared__ unsigned int lA[16 * RSA];  // bf16 pairs of a, row M
    __shared__ float        lS[16 * RSS];  // f32 s rows

    for (int t = 1; t < Tn; ++t) {
        // prefetch P(b,t) and c(b,t) early (overlaps LDS/WMMA work)
        uint4 pld[4];
        {
            const uint4* src = reinterpret_cast<const uint4*>(P + ((long)b * Tn + t) * Ln + half * 32);
            pld[0] = src[0]; pld[1] = src[1]; pld[2] = src[2]; pld[3] = src[3];
        }
        const float cbt = c[(long)b * Tn + t];

        // --- stage a as bf16 into lA row M (dword offset half*16)
        {
            unsigned int* dst = &lA[M * RSA + half * 16];
            for (int cc = 0; cc < 4; ++cc) {
                union { uint4 u; unsigned int w[4]; } pk4;
                for (int w = 0; w < 4; ++w) {
                    union { unsigned int u; __bf16 bb[2]; } pk;
                    pk.bb[0] = (__bf16)a[cc * 8 + w * 2];
                    pk.bb[1] = (__bf16)a[cc * 8 + w * 2 + 1];
                    pk4.w[w] = pk.u;
                }
                reinterpret_cast<uint4*>(dst)[cc] = pk4.u;
            }
        }
        asm volatile("s_wait_dscnt 0" ::: "memory");

        // --- gather WMMA-A fragments: lane M=l%16, K pattern per 16-bit A layout
        v16bf Af[2];
        for (int kc = 0; kc < 2; ++kc) {
            union { unsigned int w[8]; v16bf v; } fr;
            const unsigned int* rowp = &lA[M * RSA];
            for (int vv = 0; vv < 8; ++vv) {
                const int K0 = kc * 32 + half * 8 + (vv < 4 ? 2 * vv : 16 + 2 * (vv - 4));
                fr.w[vv] = rowp[K0 >> 1];
            }
            Af[kc] = fr.v;
        }

        // --- s[M][j] = sum_i a[M][i] * E[i][j]  (8 WMMAs, f32 accum)
        v8f acc[4];
        for (int nt = 0; nt < 4; ++nt) {
            v8f z = {};
            z = __builtin_amdgcn_wmma_f32_16x16x32_bf16(false, Af[0], false, Bf[0][nt],
                                                        (short)0, z, false, false);
            z = __builtin_amdgcn_wmma_f32_16x16x32_bf16(false, Af[1], false, Bf[1][nt],
                                                        (short)0, z, false, false);
            acc[nt] = z;
        }

        // --- scatter D (lane holds j = nt*16 + M, rows r+8*half) to lS
        for (int nt = 0; nt < 4; ++nt)
            for (int r = 0; r < 8; ++r)
                lS[(r + 8 * half) * RSS + nt * 16 + M] = acc[nt][r];
        asm volatile("s_wait_dscnt 0" ::: "memory");

        // --- gather own row, multiply by P, renormalize, mask, accumulate log
        float sr[32];
        {
            const uint4* rp = reinterpret_cast<const uint4*>(&lS[M * RSS + half * 32]);
            for (int cc = 0; cc < 8; ++cc) {
                union { uint4 u; float f[4]; } f4; f4.u = rp[cc];
                sr[cc*4+0]=f4.f[0]; sr[cc*4+1]=f4.f[1]; sr[cc*4+2]=f4.f[2]; sr[cc*4+3]=f4.f[3];
            }
        }
        float cand[32]; float m = 0.f;
        for (int cc = 0; cc < 4; ++cc) {
            union { uint4 u; __bf16 bv[8]; } u8; u8.u = pld[cc];
            for (int e = 0; e < 8; ++e) {
                const float v = sr[cc * 8 + e] * (float)u8.bv[e];
                cand[cc * 8 + e] = v;
                m = fmaxf(m, v);
            }
        }
        m = fmaxf(m, __shfl_xor(m, 16));
        m = fmaxf(m, 1e-30f);
        const float inv = __builtin_amdgcn_rcpf(m);
        const bool valid = (t < len);
        for (int q = 0; q < 32; ++q) a[q] = valid ? cand[q] * inv : a[q];
        logAcc = valid ? (logAcc + __logf(m) + cbt) : logAcc;
    }

    // --- log_norm[b] = logAcc + log(sum_j a[j])
    float s = 0.f;
    for (int q = 0; q < 32; ++q) s += a[q];
    s += __shfl_xor(s, 16);
    if (half == 0) lognorm[b] = logAcc + __logf(s);
}

// ---------------------------------------------------------------------------
// Kernel 5: out = sum_b (log_norm[b] - seq_score[b])
// ---------------------------------------------------------------------------
__global__ void k_final(const float* __restrict__ lognorm, const float* __restrict__ seq,
                        float* __restrict__ out) {
    __shared__ float red[256];
    const int i = threadIdx.x;
    red[i] = lognorm[i] - seq[i];
    __syncthreads();
    for (int s = 128; s > 0; s >>= 1) {
        if (i < s) red[i] += red[i + s];
        __syncthreads();
    }
    if (i == 0) out[0] = red[0];
}

// ---------------------------------------------------------------------------
extern "C" void kernel_launch(void* const* d_in, const int* in_sizes, int n_in,
                              void* d_out, int out_size, void* d_ws, size_t ws_size,
                              hipStream_t stream) {
    const float* logits = (const float*)d_in[0];
    const int*   labels = (const int*)  d_in[1];
    const int*   lens   = (const int*)  d_in[2];
    const float* trans  = (const float*)d_in[3];

    char* ws = (char*)d_ws;
    __bf16* P   = (__bf16*)(ws);                       // 32 MiB: B*T*L bf16
    float*  c   = (float*) (ws + 33554432);            // 1 MiB:  B*T f32
    __bf16* E   = (__bf16*)(ws + 34603008);            // 8 KiB:  L*L bf16
    float*  seq = (float*) (ws + 34611200);            // 1 KiB:  B f32
    float*  ln  = (float*) (ws + 34612224);            // 1 KiB:  B f32

    k_exp_rows <<<Bn * Tn, 32,  0, stream>>>(logits, P, c);
    k_exp_trans<<<1,       256, 0, stream>>>(trans, E);
    k_seq      <<<Bn,      256, 0, stream>>>(logits, labels, lens, trans, seq);
    k_scan     <<<Bn / 16, 32,  0, stream>>>(P, c, E, lens, ln);
    k_final    <<<1,       256, 0, stream>>>(ln, seq, (float*)d_out);
}
